// DependencyGNN_46093589021375
// MI455X (gfx1250) — matrix operand order repaired and checked by reading, not compiled, not run on hardware
//
#include <hip/hip_runtime.h>
#include <hip/hip_bf16.h>
#include <stddef.h>
#include <stdint.h>

#define N_NODES    200000
#define N_EDGES    400000
#define IN_CH      768
#define HID_CH     256
#define NUM_GRAPHS 8000
#define NCOLS      256            // hidden == output channels

typedef __attribute__((ext_vector_type(16))) __bf16 v16bf;
typedef __attribute__((ext_vector_type(8)))  float  v8f;

// LDS A-tile: 16 rows x 32 bf16, stored packed as 16 uints/row, padded to 20
// uints (80 B) so b128 fragment reads are 16B-aligned and bank-conflict-free.
#define LDS_STRIDE 20

// ---------------------------------------------------------------------------
// Generic fill
// ---------------------------------------------------------------------------
__global__ void fill_kernel(float* __restrict__ p, float v, size_t n) {
    size_t i = (size_t)blockIdx.x * blockDim.x + threadIdx.x;
    size_t stride = (size_t)gridDim.x * blockDim.x;
    for (; i < n; i += stride) p[i] = v;
}

// ---------------------------------------------------------------------------
// Degrees (pre-initialized to 1.0 = self loop) -> dinv
// ---------------------------------------------------------------------------
__global__ void deg_kernel(const int* __restrict__ edge, float* __restrict__ deg, int E) {
    int e = blockIdx.x * blockDim.x + threadIdx.x;
    if (e < E) atomicAdd(&deg[edge[E + e]], 1.0f);
}

__global__ void dinv_kernel(float* __restrict__ deg, int n) {
    int i = blockIdx.x * blockDim.x + threadIdx.x;
    if (i < n) deg[i] = rsqrtf(deg[i]);
}

// ---------------------------------------------------------------------------
// One-time weight pre-convert + swizzle into this file's B-fragment order (contiguous k, same as A).
// swz[(((kt*16 + nt)*32 + lane)*16 + i] = bf16( W[k*256 + n] )
//   k = kt*32 + (lane>>4)*16 + i ;  n = nt*16 + (lane&15)
// => per (kt,nt) tile, each lane's 16 bf16 are 32 contiguous bytes.
// ---------------------------------------------------------------------------
__global__ void wswz_kernel(const float* __restrict__ W, __bf16* __restrict__ swz,
                            int total /* K*256 */) {
    int tid = blockIdx.x * blockDim.x + threadIdx.x;
    if (tid >= total) return;
    int i    = tid & 15;
    int lane = (tid >> 4) & 31;
    int nt   = (tid >> 9) & 15;
    int kt   = tid >> 13;
    int k = kt * 32 + (lane >> 4) * 16 + i;
    int n = nt * 16 + (lane & 15);
    swz[tid] = (__bf16)W[(size_t)k * NCOLS + n];
}

// ---------------------------------------------------------------------------
// Fragment helpers
// ---------------------------------------------------------------------------
__device__ __forceinline__ v16bf load_afrag(const unsigned int* lds, int lane16, int half) {
    union { uint4 q[2]; v16bf v; } u;
    const unsigned int* row = lds + lane16 * LDS_STRIDE + half * 8;
    u.q[0] = *(const uint4*)(row);        // K = half*16 .. +7
    u.q[1] = *(const uint4*)(row + 4);    // K = half*16+8 .. +15
    return u.v;
}

__device__ __forceinline__ v16bf load_bfrag(const __bf16* __restrict__ swz,
                                            int kt, int nt, int lane) {
    return *(const v16bf*)(swz + (((size_t)kt * 16 + nt) * 32 + lane) * 16);
}

// ---------------------------------------------------------------------------
// GEMM layer 1 (fp32 input):  out[M x 256] = in[M x K] @ W[K x 256]
// 256 threads (8 waves) per 16-row stripe; wave owns 32 columns (2 acc tiles).
// A tile converted fp32->bf16 with v_cvt_pk on the LDS store path.
// ---------------------------------------------------------------------------
__global__ __launch_bounds__(256)
void gemm1_wmma(const float* __restrict__ in, const __bf16* __restrict__ swz,
                float* __restrict__ out, int K) {
    __shared__ unsigned int lds[16 * LDS_STRIDE];

    const int t      = threadIdx.x;
    const int wave   = t >> 5;
    const int lane   = t & 31;
    const int half   = lane >> 4;
    const int lane16 = lane & 15;
    const int m0     = blockIdx.x * 16;
    const int nt0    = wave * 2;

    v8f c0 = {}, c1 = {};

    int kt = 0;
    for (int kk = 0; kk < K; kk += 32, ++kt) {
        // stage A tile: 256 threads x one bf16-pair each (coalesced b64 loads)
        {
            int row = t >> 4, cp = t & 15;
            float2 f = *(const float2*)(in + (size_t)(m0 + row) * K + kk + cp * 2);
            union { __bf16 h[2]; unsigned int u; } pk;
            pk.h[0] = (__bf16)f.x;
            pk.h[1] = (__bf16)f.y;
            lds[row * LDS_STRIDE + cp] = pk.u;
        }
        __syncthreads();

        v16bf a  = load_afrag(lds, lane16, half);
        v16bf b0 = load_bfrag(swz, kt, nt0,     lane);
        v16bf b1 = load_bfrag(swz, kt, nt0 + 1, lane);

        c0 = __builtin_amdgcn_wmma_f32_16x16x32_bf16(false, a, false, b0,
                                                     (short)0, c0, false, false);
        c1 = __builtin_amdgcn_wmma_f32_16x16x32_bf16(false, a, false, b1,
                                                     (short)0, c1, false, false);
        __syncthreads();
    }

#pragma unroll
    for (int r = 0; r < 8; ++r) {
        size_t off = (size_t)(m0 + r + half * 8) * NCOLS + nt0 * 16 + lane16;
        out[off]      = c0[r];
        out[off + 16] = c1[r];
    }
}

// ---------------------------------------------------------------------------
// GEMM layer 2 (bf16 input, K=256): A tile staged by the CDNA5 async-to-LDS
// DMA path (global_load_async_to_lds_b128, ASYNCcnt-tracked) — 2 waves move
// the 1 KB tile, 16 B per lane, no VGPR round-trip, no conversion needed.
// ---------------------------------------------------------------------------
__global__ __launch_bounds__(256)
void gemm2_wmma(const __bf16* __restrict__ in16, const __bf16* __restrict__ swz,
                float* __restrict__ out) {
    __shared__ unsigned int lds[16 * LDS_STRIDE];

    const int t      = threadIdx.x;
    const int wave   = t >> 5;
    const int lane   = t & 31;
    const int half   = lane >> 4;
    const int lane16 = lane & 15;
    const int m0     = blockIdx.x * 16;
    const int nt0    = wave * 2;

    v8f c0 = {}, c1 = {};

    int kt = 0;
    for (int kk = 0; kk < HID_CH; kk += 32, ++kt) {
        if (t < 64) {   // waves 0..1: async-DMA the 16x32 bf16 tile into LDS
            int row = t >> 2, seg = t & 3;               // 4 x 16B segments/row
            const __bf16* g = in16 + (size_t)(m0 + row) * NCOLS + kk + seg * 8;
            unsigned long long ga = (unsigned long long)(uintptr_t)g;
            unsigned int la = (unsigned int)(uintptr_t)(void*)
                                  &lds[row * LDS_STRIDE + seg * 4];
            asm volatile("global_load_async_to_lds_b128 %0, %1, off"
                         :: "v"(la), "v"(ga) : "memory");
            asm volatile("s_wait_asynccnt 0x0" ::: "memory");
        }
        __syncthreads();

        v16bf a  = load_afrag(lds, lane16, half);
        v16bf b0 = load_bfrag(swz, kt, nt0,     lane);
        v16bf b1 = load_bfrag(swz, kt, nt0 + 1, lane);

        c0 = __builtin_amdgcn_wmma_f32_16x16x32_bf16(false, a, false, b0,
                                                     (short)0, c0, false, false);
        c1 = __builtin_amdgcn_wmma_f32_16x16x32_bf16(false, a, false, b1,
                                                     (short)0, c1, false, false);
        __syncthreads();
    }

#pragma unroll
    for (int r = 0; r < 8; ++r) {
        size_t off = (size_t)(m0 + r + half * 8) * NCOLS + nt0 * 16 + lane16;
        out[off]      = c0[r];
        out[off + 16] = c1[r];
    }
}

// ---------------------------------------------------------------------------
// Edge aggregation: one wave per edge, lane owns 8 contiguous channels
// (two b128 gathers + 8 f32 atomics): B[dst,:] += A[src,:]*dinv[src]*dinv[dst]
// ---------------------------------------------------------------------------
__global__ __launch_bounds__(256)
void agg_edges_kernel(const float* __restrict__ A, float* __restrict__ B,
                      const int* __restrict__ edge, const float* __restrict__ dinv,
                      int E) {
    int e = blockIdx.x * 8 + (threadIdx.x >> 5);
    if (e >= E) return;
    int lane = threadIdx.x & 31;
    int src = edge[e];
    int dst = edge[E + e];
    float norm = dinv[src] * dinv[dst];
    const float4* a4 = (const float4*)(A + (size_t)src * NCOLS + lane * 8);
    float*        bp = B + (size_t)dst * NCOLS + lane * 8;
    float4 x0 = a4[0], x1 = a4[1];
    atomicAdd(bp + 0, x0.x * norm); atomicAdd(bp + 1, x0.y * norm);
    atomicAdd(bp + 2, x0.z * norm); atomicAdd(bp + 3, x0.w * norm);
    atomicAdd(bp + 4, x1.x * norm); atomicAdd(bp + 5, x1.y * norm);
    atomicAdd(bp + 6, x1.z * norm); atomicAdd(bp + 7, x1.w * norm);
}

// ---------------------------------------------------------------------------
// Fold self-loop + bias (+ReLU). If out16 != null: write bf16 (feeds GEMM2,
// halving its read traffic); else update B in fp32 (final features).
//   v = B[i,c] + dinv[i]^2 * A[i,c] + bias[c]
// ---------------------------------------------------------------------------
__global__ __launch_bounds__(256)
void selfloop_bias_act_kernel(const float* __restrict__ A, float* __restrict__ B,
                              __bf16* __restrict__ out16,
                              const float* __restrict__ dinv,
                              const float* __restrict__ bias, int do_relu) {
    size_t idx = (size_t)blockIdx.x * 256 + threadIdx.x;
    int i = (int)(idx >> 8);
    int c = (int)(idx & 255);
    float d = dinv[i];
    float v = B[idx] + d * d * A[idx] + bias[c];
    if (do_relu) v = fmaxf(v, 0.0f);
    if (out16) out16[idx] = (__bf16)v;
    else       B[idx] = v;
}

// ---------------------------------------------------------------------------
// Pooling
// ---------------------------------------------------------------------------
__global__ void counts_kernel(const int* __restrict__ batch, float* __restrict__ counts,
                              int n) {
    int i = blockIdx.x * blockDim.x + threadIdx.x;
    if (i < n) atomicAdd(&counts[batch[i]], 1.0f);
}

__global__ __launch_bounds__(256)
void pool_scatter_kernel(const float* __restrict__ B, const int* __restrict__ batch,
                         float* __restrict__ out, int n) {
    int i = blockIdx.x * 8 + (threadIdx.x >> 5);
    if (i >= n) return;
    int lane = threadIdx.x & 31;
    int g = batch[i];
    const float4* b4 = (const float4*)(B + (size_t)i * NCOLS + lane * 8);
    float*        op = out + (size_t)g * NCOLS + lane * 8;
    float4 x0 = b4[0], x1 = b4[1];
    atomicAdd(op + 0, x0.x); atomicAdd(op + 1, x0.y);
    atomicAdd(op + 2, x0.z); atomicAdd(op + 3, x0.w);
    atomicAdd(op + 4, x1.x); atomicAdd(op + 5, x1.y);
    atomicAdd(op + 6, x1.z); atomicAdd(op + 7, x1.w);
}

__global__ void normalize_kernel(float* __restrict__ out, const float* __restrict__ counts,
                                 int total) {
    int idx = blockIdx.x * blockDim.x + threadIdx.x;
    if (idx < total) out[idx] /= fmaxf(counts[idx >> 8], 1.0f);
}

// ---------------------------------------------------------------------------
// Launcher
// ---------------------------------------------------------------------------
extern "C" void kernel_launch(void* const* d_in, const int* in_sizes, int n_in,
                              void* d_out, int out_size, void* d_ws, size_t ws_size,
                              hipStream_t stream) {
    const float* x     = (const float*)d_in[0];
    const float* W1    = (const float*)d_in[1];
    const float* b1    = (const float*)d_in[2];
    const float* W2    = (const float*)d_in[3];
    const float* b2    = (const float*)d_in[4];
    const int*   edge  = (const int*)d_in[5];
    const int*   batch = (const int*)d_in[6];

    const int E = in_sizes[5] / 2;        // 400000
    const int N = in_sizes[6];            // 200000 nodes
    const size_t NM = (size_t)N * NCOLS;  // 51.2M elements

    float*  bufA   = (float*)d_ws;                    // GEMM outputs (fp32)
    float*  bufB   = bufA + NM;                       // aggregation buffer
    float*  dinv   = bufB + NM;                       // N floats
    float*  counts = dinv + N;                        // NUM_GRAPHS floats
    __bf16* W1swz  = (__bf16*)(counts + NUM_GRAPHS);  // 768*256 bf16 (swizzled)
    __bf16* W2swz  = W1swz + (size_t)IN_CH * NCOLS;   // 256*256 bf16
    __bf16* h1bf16 = W2swz + (size_t)HID_CH * NCOLS;  // N*256 bf16 (layer-1 out)
    float*  outp   = (float*)d_out;

    const int TB = 256;

    // degrees -> dinv
    fill_kernel<<<1024, TB, 0, stream>>>(dinv, 1.0f, (size_t)N);
    deg_kernel<<<(E + TB - 1) / TB, TB, 0, stream>>>(edge, dinv, E);
    dinv_kernel<<<(N + TB - 1) / TB, TB, 0, stream>>>(dinv, N);

    // weight pre-convert + swizzle (one-time, L2-resident afterwards)
    wswz_kernel<<<(IN_CH * NCOLS) / TB, TB, 0, stream>>>(W1, W1swz, IN_CH * NCOLS);
    wswz_kernel<<<(HID_CH * NCOLS) / TB, TB, 0, stream>>>(W2, W2swz, HID_CH * NCOLS);

    // ---- layer 1 ----
    gemm1_wmma<<<N / 16, TB, 0, stream>>>(x, W1swz, bufA, IN_CH);
    fill_kernel<<<4096, TB, 0, stream>>>(bufB, 0.0f, NM);
    agg_edges_kernel<<<(E + 7) / 8, TB, 0, stream>>>(bufA, bufB, edge, dinv, E);
    selfloop_bias_act_kernel<<<(unsigned)(NM / TB), TB, 0, stream>>>(
        bufA, bufB, h1bf16, dinv, b1, 1);

    // ---- layer 2 ----
    gemm2_wmma<<<N / 16, TB, 0, stream>>>(h1bf16, W2swz, bufA);
    fill_kernel<<<4096, TB, 0, stream>>>(bufB, 0.0f, NM);
    agg_edges_kernel<<<(E + 7) / 8, TB, 0, stream>>>(bufA, bufB, edge, dinv, E);
    selfloop_bias_act_kernel<<<(unsigned)(NM / TB), TB, 0, stream>>>(
        bufA, bufB, (__bf16*)nullptr, dinv, b2, 0);

    // ---- global mean pool ----
    const int OUT_TOTAL = NUM_GRAPHS * NCOLS;
    fill_kernel<<<1024, TB, 0, stream>>>(outp, 0.0f, (size_t)OUT_TOTAL);
    fill_kernel<<<64, TB, 0, stream>>>(counts, 0.0f, (size_t)NUM_GRAPHS);
    counts_kernel<<<(N + TB - 1) / TB, TB, 0, stream>>>(batch, counts, N);
    pool_scatter_kernel<<<(N + 7) / 8, TB, 0, stream>>>(bufB, batch, outp, N);
    normalize_kernel<<<(OUT_TOTAL + TB - 1) / TB, TB, 0, stream>>>(outp, counts, OUT_TOTAL);
}
